// Attention_32323923870437
// MI455X (gfx1250) — matrix operand order repaired and checked by reading, not compiled, not run on hardware
//
#include <hip/hip_runtime.h>
#include <hip/hip_bf16.h>

// ---------------------------------------------------------------------------
// SAGAN non-local attention block for MI455X (gfx1250), bf16 WMMA pipeline.
// Shapes: B=16, C=256, H=W=64, N=4096, M=1024, C/8=32, C/2=128.
// ---------------------------------------------------------------------------

typedef __attribute__((ext_vector_type(16))) __bf16 v16bf;
typedef __attribute__((ext_vector_type(8)))  __bf16 v8bf;
typedef __attribute__((ext_vector_type(4)))  __bf16 v4bf;
typedef __attribute__((ext_vector_type(8)))  float  v8f;

#define NB   16
#define CC   256
#define NN   4096
#define MM   1024
#define C8   32
#define C2   128
#define OC   192   // concatenated projection rows (32 theta + 32 phi + 128 g)
#define OF   160   // full-res phi+g rows

static __device__ __forceinline__ v16bf load_ab16(const __bf16* lo, const __bf16* hi) {
    v16bf r;
    *(v8bf*)&r        = *(const v8bf*)lo;
    ((v8bf*)&r)[1]    = *(const v8bf*)hi;
    return r;
}

static __device__ __forceinline__ v8f zero8() {
    v8f z = {0.f, 0.f, 0.f, 0.f, 0.f, 0.f, 0.f, 0.f};
    return z;
}

// ---------------------------------------------------------------------------
// Kernel 0: cast weights to bf16.  Wcat[o<192][c<256] = theta|phi|g rows.
// Wo_bf[co<256][ci<128].
// ---------------------------------------------------------------------------
__global__ void __launch_bounds__(256)
k_prep(const float* __restrict__ wt, const float* __restrict__ wp,
       const float* __restrict__ wg, const float* __restrict__ wo,
       __bf16* __restrict__ Wcat, __bf16* __restrict__ Wo_bf)
{
    int i = blockIdx.x * 256 + threadIdx.x;
    if (i < OC * CC) {
        int o = i >> 8, c = i & 255;
        float v = (o < 32) ? wt[o * CC + c]
                : (o < 64) ? wp[(o - 32) * CC + c]
                           : wg[(o - 64) * CC + c];
        Wcat[i] = (__bf16)v;
    }
    if (i < CC * C2) Wo_bf[i] = (__bf16)wo[i];
}

// ---------------------------------------------------------------------------
// Kernel 1: projections.  Per WG: one batch, 64 consecutive n.
// D[o, n] = Wcat[o, c] * x[c, n]  (K = 256).
// A = Wcat rows (contiguous bf16 from global), B = x tile (LDS, bf16).
// theta rows (o<32) -> theta_t[b][n][32]; rest -> yfull[b][o-32][n].
// ---------------------------------------------------------------------------
__global__ void __launch_bounds__(256)
k_proj(const float* __restrict__ x, const __bf16* __restrict__ Wcat,
       __bf16* __restrict__ theta_t, __bf16* __restrict__ yfull)
{
    __shared__ __align__(16) __bf16 Xn[CC * 72];   // [c][64 n + pad]

    const int b  = blockIdx.x >> 6;
    const int n0 = (blockIdx.x & 63) << 6;
    const int tid = threadIdx.x;

    // Vectorized stage-in: float4 global loads, packed b64 LDS stores.
    const float* xb = x + (size_t)b * CC * NN + n0;
    for (int i = tid * 4; i < CC * 64; i += 256 * 4) {
        int c = i >> 6, w = i & 63;
        float4 f = *(const float4*)(xb + (size_t)c * NN + w);
        v4bf p;
        p[0] = (__bf16)f.x; p[1] = (__bf16)f.y;
        p[2] = (__bf16)f.z; p[3] = (__bf16)f.w;
        *(v4bf*)&Xn[c * 72 + w] = p;
    }
    __syncthreads();

    const int wv   = tid >> 5;
    const int lane = tid & 31;
    const int col  = lane & 15;
    const int kh   = lane >> 4;
    const int ntile = wv & 3;            // which 16-n slice of the 64
    const int og0   = (wv >> 2) * 6;     // 6 o-tiles per wave (12 total)

    v8f acc[6];
#pragma unroll
    for (int t = 0; t < 6; ++t) acc[t] = zero8();

    const int ncol = ntile * 16 + col;
    for (int kc = 0; kc < 8; ++kc) {     // K chunks of 32 over C=256
        // B frag: col = ncol, K = kc*32 + kh*16 + i  (gather from LDS)
        v16bf bfrag;
#pragma unroll
        for (int i = 0; i < 16; ++i)
            bfrag[i] = Xn[(kc * 32 + kh * 16 + i) * 72 + ncol];
#pragma unroll
        for (int t = 0; t < 6; ++t) {
            int o = (og0 + t) * 16 + col;                 // A row
            const __bf16* ap = Wcat + (size_t)o * CC + kc * 32 + kh * 16;
            v16bf afrag = load_ab16(ap, ap + 8);
            acc[t] = __builtin_amdgcn_wmma_f32_16x16x32_bf16(
                false, afrag, false, bfrag, (short)0, acc[t], false, false);
        }
    }

    const int n = n0 + ntile * 16 + col;                  // D col
#pragma unroll
    for (int t = 0; t < 6; ++t) {
#pragma unroll
        for (int j = 0; j < 8; ++j) {
            int o = (og0 + t) * 16 + j + 8 * kh;          // D row
            float v = acc[t][j];
            if (o < 32)
                theta_t[((size_t)(b * NN + n)) * C8 + o] = (__bf16)v;
            else
                yfull[((size_t)(b * OF + (o - 32))) * NN + n] = (__bf16)v;
        }
    }
}

// ---------------------------------------------------------------------------
// Kernel 1b: 2x2 max pool on full-res phi/g.
// phi rows (o<32) -> phi_t[b][m][32] (B-frag friendly);
// g rows -> g_cm[b][c][m] (contiguous along m).
// ---------------------------------------------------------------------------
__global__ void __launch_bounds__(256)
k_pool(const __bf16* __restrict__ yfull,
       __bf16* __restrict__ phi_t, __bf16* __restrict__ g_cm)
{
    int idx = blockIdx.x * 256 + threadIdx.x;            // b*OF*M total
    if (idx >= NB * OF * MM) return;
    int m = idx & (MM - 1);
    int o = (idx >> 10) % OF;
    int b = idx / (OF * MM);

    const __bf16* src = yfull + ((size_t)(b * OF + o)) * NN
                        + ((m >> 5) * 128 + (m & 31) * 2);
    float v0 = (float)src[0],  v1 = (float)src[1];
    float v2 = (float)src[64], v3 = (float)src[65];
    float v = fmaxf(fmaxf(v0, v1), fmaxf(v2, v3));

    if (o < 32) phi_t[((size_t)(b * MM + m)) * C8 + o] = (__bf16)v;
    else        g_cm[((size_t)(b * C2 + (o - 32))) * MM + m] = (__bf16)v;
}

// ---------------------------------------------------------------------------
// Kernel 2: fused flash attention + output projection + residual.
// Grid: B * (N/64).  Block: 128 threads (4 waves); wave w owns 16 rows of n.
// Streams M=1024 in chunks of 64 with running-max softmax; scores never hit
// global memory.  Then O2 = (O/l) * Wo^T, out = gamma*O2 + x.
// ---------------------------------------------------------------------------
__global__ void __launch_bounds__(128)
k_attn(const float* __restrict__ x, const __bf16* __restrict__ theta_t,
       const __bf16* __restrict__ phi_t, const __bf16* __restrict__ g_cm,
       const __bf16* __restrict__ Wo, const float* __restrict__ gamma_p,
       float* __restrict__ out)
{
    __shared__ __align__(16) __bf16 Pl[4 * 16 * 72];    // per-wave P staging
    __shared__ __align__(16) __bf16 Ol[4 * 16 * 136];   // per-wave O staging
    __shared__ __align__(16) __bf16 O2l[CC * 72];       // WG output staging

    const int b  = blockIdx.x >> 6;
    const int n0 = (blockIdx.x & 63) << 6;
    const int tid  = threadIdx.x;
    const int wv   = tid >> 5;
    const int lane = tid & 31;
    const int col  = lane & 15;
    const int kh   = lane >> 4;

    __bf16* pl = Pl + wv * 16 * 72;
    __bf16* ol = Ol + wv * 16 * 136;
    const int nb = n0 + wv * 16;

    // A_theta: 16x32 frag, rows = n, K = 32 channels (whole contraction)
    const __bf16* tp = theta_t + ((size_t)(b * NN + nb + col)) * C8 + kh * 16;
    v16bf a_th = load_ab16(tp, tp + 8);

    v8f acc[8];
#pragma unroll
    for (int t = 0; t < 8; ++t) acc[t] = zero8();
    float rmax[8], rsum[8];
#pragma unroll
    for (int j = 0; j < 8; ++j) { rmax[j] = -1e30f; rsum[j] = 0.f; }

    const v8f zc = zero8();

    for (int mc = 0; mc < 16; ++mc) {                   // M chunks of 64
        const int m0 = mc * 64;

        // ---- scores: S[16 x 64] = theta_tile @ phi ----
        v8f S[4];
#pragma unroll
        for (int ct = 0; ct < 4; ++ct) {
            const __bf16* pp = phi_t
                + ((size_t)(b * MM + m0 + ct * 16 + col)) * C8 + kh * 16;
            v16bf bphi = *(const v16bf*)pp;
            S[ct] = __builtin_amdgcn_wmma_f32_16x16x32_bf16(
                false, a_th, false, bphi, (short)0, zc, false, false);
        }

        // ---- streaming softmax (row = j + 8*kh, cols across 16 lanes) ----
#pragma unroll
        for (int j = 0; j < 8; ++j) {
            float cm = fmaxf(fmaxf(S[0][j], S[1][j]), fmaxf(S[2][j], S[3][j]));
            cm = fmaxf(cm, __shfl_xor(cm, 1));
            cm = fmaxf(cm, __shfl_xor(cm, 2));
            cm = fmaxf(cm, __shfl_xor(cm, 4));
            cm = fmaxf(cm, __shfl_xor(cm, 8));
            float nm = fmaxf(rmax[j], cm);
            float sc = __expf(rmax[j] - nm);
            float ss = 0.f;
#pragma unroll
            for (int ct = 0; ct < 4; ++ct) {
                float p = __expf(S[ct][j] - nm);
                S[ct][j] = p;
                ss += p;
            }
            ss += __shfl_xor(ss, 1);
            ss += __shfl_xor(ss, 2);
            ss += __shfl_xor(ss, 4);
            ss += __shfl_xor(ss, 8);
            rsum[j] = rsum[j] * sc + ss;
            rmax[j] = nm;
#pragma unroll
            for (int ct8 = 0; ct8 < 8; ++ct8) acc[ct8][j] *= sc;
        }

        // ---- P (C/D layout) -> LDS -> A-frag layout ----
#pragma unroll
        for (int ct = 0; ct < 4; ++ct)
#pragma unroll
            for (int j = 0; j < 8; ++j)
                pl[(j + 8 * kh) * 72 + ct * 16 + col] = (__bf16)S[ct][j];
        __syncthreads();   // uniform; orders DS writes before reads

        // ---- O += P @ g^T  (K = 64 over m-chunk) ----
#pragma unroll
        for (int af = 0; af < 2; ++af) {
            const __bf16* pa = pl + col * 72 + af * 32 + kh * 16;
            v16bf a_p = load_ab16(pa, pa + 8);
#pragma unroll
            for (int ct8 = 0; ct8 < 8; ++ct8) {
                const __bf16* gp = g_cm
                    + ((size_t)(b * C2 + ct8 * 16 + col)) * MM
                    + m0 + af * 32 + kh * 16;
                v16bf bg = *(const v16bf*)gp;
                acc[ct8] = __builtin_amdgcn_wmma_f32_16x16x32_bf16(
                    false, a_p, false, bg, (short)0, acc[ct8], false, false);
            }
        }
        __syncthreads();
    }

    // ---- normalize and stage O (bf16) for projection A-frags ----
    float inv[8];
#pragma unroll
    for (int j = 0; j < 8; ++j) inv[j] = 1.f / rsum[j];
#pragma unroll
    for (int ct8 = 0; ct8 < 8; ++ct8)
#pragma unroll
        for (int j = 0; j < 8; ++j)
            ol[(j + 8 * kh) * 136 + ct8 * 16 + col] =
                (__bf16)(acc[ct8][j] * inv[j]);
    __syncthreads();

    // ---- O2[16 x 256] = O[16 x 128] @ Wo^T ----
    v8f accw[16];
#pragma unroll
    for (int t = 0; t < 16; ++t) accw[t] = zero8();
    for (int kk = 0; kk < 4; ++kk) {                    // K chunks of 32 over 128
        const __bf16* oa = ol + col * 136 + kk * 32 + kh * 16;
        v16bf a_o = load_ab16(oa, oa + 8);
#pragma unroll
        for (int cot = 0; cot < 16; ++cot) {
            const __bf16* wpp = Wo + (size_t)(cot * 16 + col) * C2
                                + kk * 32 + kh * 16;
            v16bf bw = *(const v16bf*)wpp;
            accw[cot] = __builtin_amdgcn_wmma_f32_16x16x32_bf16(
                false, a_o, false, bw, (short)0, accw[cot], false, false);
        }
    }

    // ---- stage O2 in LDS for coalesced residual epilogue ----
    __syncthreads();  // all Pl/Ol reads done before O2l phase
#pragma unroll
    for (int cot = 0; cot < 16; ++cot)
#pragma unroll
        for (int j = 0; j < 8; ++j)
            O2l[(cot * 16 + col) * 72 + wv * 16 + j + 8 * kh] =
                (__bf16)accw[cot][j];
    __syncthreads();

    // ---- residual epilogue: float4 coalesced read-modify-write ----
    const float gamma = gamma_p[0];
    const size_t xbase = (size_t)b * CC * NN + n0;
    for (int i = tid * 4; i < CC * 64; i += 128 * 4) {
        int co = i >> 6, wc = i & 63;
        size_t ad = xbase + (size_t)co * NN + wc;
        float4 xv = *(const float4*)(x + ad);
        v4bf o4 = *(const v4bf*)&O2l[co * 72 + wc];
        float4 ov;
        ov.x = gamma * (float)o4[0] + xv.x;
        ov.y = gamma * (float)o4[1] + xv.y;
        ov.z = gamma * (float)o4[2] + xv.z;
        ov.w = gamma * (float)o4[3] + xv.w;
        *(float4*)(out + ad) = ov;
    }
}

// ---------------------------------------------------------------------------
// Launch
// ---------------------------------------------------------------------------
extern "C" void kernel_launch(void* const* d_in, const int* in_sizes, int n_in,
                              void* d_out, int out_size, void* d_ws, size_t ws_size,
                              hipStream_t stream)
{
    const float* x       = (const float*)d_in[0];
    const float* w_theta = (const float*)d_in[1];
    const float* w_phi   = (const float*)d_in[2];
    const float* w_g     = (const float*)d_in[3];
    const float* w_o     = (const float*)d_in[4];
    const float* gamma   = (const float*)d_in[5];

    char* ws = (char*)d_ws;
    // byte offsets (256-aligned)
    __bf16* Wcat    = (__bf16*)(ws + 0);         //  98,304 B
    __bf16* Wo_bf   = (__bf16*)(ws + 98304);     //  65,536 B
    __bf16* theta_t = (__bf16*)(ws + 163840);    //  4,194,304 B
    __bf16* phi_t   = (__bf16*)(ws + 4358144);   //  1,048,576 B
    __bf16* g_cm    = (__bf16*)(ws + 5406720);   //  4,194,304 B
    __bf16* yfull   = (__bf16*)(ws + 9601024);   // 20,971,520 B  (total ~29.2 MB)

    k_prep<<<192, 256, 0, stream>>>(w_theta, w_phi, w_g, w_o, Wcat, Wo_bf);
    k_proj<<<NB * (NN / 64), 256, 0, stream>>>(x, Wcat, theta_t, yfull);
    k_pool<<<(NB * OF * MM) / 256, 256, 0, stream>>>(yfull, phi_t, g_cm);
    k_attn<<<NB * (NN / 64), 128, 0, stream>>>(x, theta_t, phi_t, g_cm,
                                               Wo_bf, gamma, (float*)d_out);
}
